// GCN_40020505264473
// MI455X (gfx1250) — matrix operand order repaired and checked by reading, not compiled, not run on hardware
//
#include <hip/hip_runtime.h>
#include <hip/hip_bf16.h>

// GraphSAGE forward for MI455X (gfx1250).
// - Edge aggregation via global_atomic_add_f32 (all activation buffers fit in
//   the 192MB L2, so the random scatter stays on-chip).
// - GEMMs via v_wmma_f32_16x16x32_bf16, register-blocked 16x(16*NT) strip per
//   wave, ping-pong double buffering (2 K-steps per loop iteration) so there
//   are no inter-buffer register copies and WMMAs overlap in-flight loads.

typedef __attribute__((ext_vector_type(16))) __bf16 v16bf;
typedef __attribute__((ext_vector_type(8)))  __bf16 bf16x8;
typedef __attribute__((ext_vector_type(8)))  float  floatx8;

#define NNODES 100000
#define NEDGES 3200000
#define NGRAPH 256

// ---------------- small utility kernels ----------------

__global__ void count_kernel(const int* __restrict__ idx, float* __restrict__ cnt, int n) {
  int i = blockIdx.x * blockDim.x + threadIdx.x;
  if (i < n) atomicAdd(&cnt[idx[i]], 1.0f);
}

__global__ void recip_clamp_kernel(const float* __restrict__ d, float* __restrict__ out, int n) {
  int i = blockIdx.x * blockDim.x + threadIdx.x;
  if (i < n) out[i] = 1.0f / fmaxf(d[i], 1.0f);
}

// fp32 (rows_in x cols_in) -> zero-padded bf16 (rows_out x cols_out), row-major
__global__ void cvt_pad_kernel(const float* __restrict__ in, __bf16* __restrict__ out,
                               int rows_out, int cols_out, int rows_in, int cols_in) {
  long long i = blockIdx.x * (long long)blockDim.x + threadIdx.x;
  long long total = (long long)rows_out * cols_out;
  if (i >= total) return;
  int r = (int)(i / cols_out), c = (int)(i % cols_out);
  float v = (r < rows_in && c < cols_in) ? in[(size_t)r * cols_in + c] : 0.0f;
  out[i] = (__bf16)v;
}

// per-(edge, 8-feature chunk) scatter-add of bf16 features into fp32 agg
__global__ void scatter_kernel(const __bf16* __restrict__ X, const int* __restrict__ src,
                               const int* __restrict__ dst, float* __restrict__ agg,
                               int nEdges, int K) {
  long long t = blockIdx.x * (long long)blockDim.x + threadIdx.x;
  int chunks = K >> 3;
  long long total = (long long)nEdges * chunks;
  if (t >= total) return;
  int e = (int)(t / chunks), c = (int)(t % chunks);
  int s = src[e], d = dst[e];
  bf16x8 v = *(const bf16x8*)(X + (size_t)s * K + (size_t)c * 8);
  float* ap = agg + (size_t)d * K + (size_t)c * 8;
#pragma unroll
  for (int i = 0; i < 8; ++i) atomicAdd(ap + i, (float)v[i]);
}

// agg[m][k] * invdeg[m] -> bf16
__global__ void scale_cvt_kernel(const float* __restrict__ agg, const float* __restrict__ invdeg,
                                 __bf16* __restrict__ out, long long total, int K) {
  long long i = blockIdx.x * (long long)blockDim.x + threadIdx.x;
  if (i >= total) return;
  int r = (int)(i / K);
  out[i] = (__bf16)(agg[i] * invdeg[r]);
}

// per-(node, 8-feature chunk) scatter-add into per-graph pool
__global__ void pool_scatter_kernel(const __bf16* __restrict__ H, const int* __restrict__ batch,
                                    float* __restrict__ pooled, int nNodes, int K) {
  long long t = blockIdx.x * (long long)blockDim.x + threadIdx.x;
  int chunks = K >> 3;
  long long total = (long long)nNodes * chunks;
  if (t >= total) return;
  int n = (int)(t / chunks), c = (int)(t % chunks);
  int g = batch[n];
  bf16x8 v = *(const bf16x8*)(H + (size_t)n * K + (size_t)c * 8);
  float* pp = pooled + (size_t)g * K + (size_t)c * 8;
#pragma unroll
  for (int i = 0; i < 8; ++i) atomicAdd(pp + i, (float)v[i]);
}

__global__ void pool_fin_kernel(const float* __restrict__ pooled, const float* __restrict__ cnt,
                                __bf16* __restrict__ out, int G, int K) {
  long long i = blockIdx.x * (long long)blockDim.x + threadIdx.x;
  long long total = (long long)G * K;
  if (i >= total) return;
  int r = (int)(i / K);
  out[i] = (__bf16)(pooled[i] / fmaxf(cnt[r], 1.0f));
}

// ---------------- WMMA GEMM ----------------
// C[MxN] = act( A1[MxK1] @ B1[NxK1]^T + A2[MxK2] @ B2[NxK2]^T + bias[N] )
// Each wave computes a 16 x (16*NT) strip. K must be a multiple of 64.
// ISA layouts (CDNA5 ISA 7.12.2):
//   A: lane(row=lane&15, half=lane>>4) -> K = {k0+8h..+8} ++ {k0+16+8h..+8}
//   B: lane(col=lane&15, half)        -> K = {k0+8h..+8} ++ {k0+16+8h..+8} of W[col][.]
//   C: VGPR r -> (M = r+8*half, N = col)

__device__ __forceinline__ v16bf load_a_frag(const __bf16* arow, int k0, int half) {
  const __bf16* p = arow + k0 + half * 8;
  bf16x8 lo = *(const bf16x8*)(p);
  bf16x8 hi = *(const bf16x8*)(p + 16);
  v16bf a;
#pragma unroll
  for (int i = 0; i < 8; ++i) { a[i] = lo[i]; a[i + 8] = hi[i]; }
  return a;
}

__device__ __forceinline__ v16bf load_b_frag(const __bf16* brow, int k0, int half) {
  const __bf16* p = brow + k0 + half * 8;
  bf16x8 lo = *(const bf16x8*)(p);
  bf16x8 hi = *(const bf16x8*)(p + 16);
  v16bf b;
#pragma unroll
  for (int i = 0; i < 8; ++i) { b[i] = lo[i]; b[i + 8] = hi[i]; }
  return b;
}

template <int NT>
__device__ __forceinline__ void load_frags(const __bf16* arow, const __bf16* const (&brow)[NT],
                                           int k0, int half, v16bf& a, v16bf (&b)[NT]) {
  a = load_a_frag(arow, k0, half);
#pragma unroll
  for (int t = 0; t < NT; ++t) b[t] = load_b_frag(brow[t], k0, half);
}

template <int NT>
__device__ __forceinline__ void do_wmma(const v16bf& a, const v16bf (&b)[NT], floatx8 (&acc)[NT]) {
#pragma unroll
  for (int t = 0; t < NT; ++t)
    acc[t] = __builtin_amdgcn_wmma_f32_16x16x32_bf16(false, a, false, b[t],
                                                     (short)0, acc[t], false, false);
}

template <int NT>
__device__ __forceinline__ void gemm_accum_pipelined(
    const __bf16* __restrict__ A, const __bf16* __restrict__ B, int K,
    int m0, int n0, int col, int half, floatx8 (&acc)[NT]) {
  const __bf16* arow = A + (size_t)(m0 + col) * K;
  const __bf16* brow[NT];
#pragma unroll
  for (int t = 0; t < NT; ++t) brow[t] = B + (size_t)(n0 + t * 16 + col) * K;

  v16bf a0, a1;
  v16bf b0[NT], b1[NT];
  load_frags<NT>(arow, brow, 0, half, a0, b0);
  int k = 0;
#pragma unroll 1
  for (; k + 64 < K; k += 64) {
    load_frags<NT>(arow, brow, k + 32, half, a1, b1);
    do_wmma<NT>(a0, b0, acc);
    load_frags<NT>(arow, brow, k + 64, half, a0, b0);
    do_wmma<NT>(a1, b1, acc);
  }
  // tail pair (K is a multiple of 64)
  load_frags<NT>(arow, brow, k + 32, half, a1, b1);
  do_wmma<NT>(a0, b0, acc);
  do_wmma<NT>(a1, b1, acc);
}

template <int NT>
__global__ __launch_bounds__(256) void wmma_gemm_kernel(
    const __bf16* __restrict__ A1, const __bf16* __restrict__ B1, int K1,
    const __bf16* __restrict__ A2, const __bf16* __restrict__ B2, int K2,
    const float* __restrict__ bias,
    __bf16* __restrict__ Cb, float* __restrict__ Cf,
    int M, int N, int ldc, int nvalid, int act) {
  int wave = (int)((blockIdx.x * (long long)blockDim.x + threadIdx.x) >> 5);
  int lane = threadIdx.x & 31;
  int ngroups = N / (16 * NT);
  int mtile = wave / ngroups;
  int ngrp = wave - mtile * ngroups;
  if (mtile * 16 >= M) return;
  int col = lane & 15;
  int half = lane >> 4;
  int m0 = mtile << 4;
  int n0 = ngrp * 16 * NT;

  floatx8 acc[NT];
#pragma unroll
  for (int t = 0; t < NT; ++t) acc[t] = (floatx8){};

  gemm_accum_pipelined<NT>(A1, B1, K1, m0, n0, col, half, acc);
  if (A2) gemm_accum_pipelined<NT>(A2, B2, K2, m0, n0, col, half, acc);

#pragma unroll
  for (int t = 0; t < NT; ++t) {
    int nn = n0 + t * 16 + col;
    float bv = (bias && nn < nvalid) ? bias[nn] : 0.0f;
#pragma unroll
    for (int r = 0; r < 8; ++r) {
      float v = acc[t][r] + bv;
      if (act == 1) v = fmaxf(v, 0.0f);
      else if (act == 2) v = (v > 0.0f) ? v : 0.01f * v;
      if (nn < nvalid) {
        size_t o = (size_t)(m0 + r + half * 8) * ldc + nn;
        if (Cb) Cb[o] = (__bf16)v;
        if (Cf) Cf[o] = v;
      }
    }
  }
}

// ---------------- host orchestration ----------------

static inline unsigned blks(long long n, int t) { return (unsigned)((n + t - 1) / t); }

extern "C" void kernel_launch(void* const* d_in, const int* in_sizes, int n_in,
                              void* d_out, int out_size, void* d_ws, size_t ws_size,
                              hipStream_t stream) {
  const float* x      = (const float*)d_in[0];
  const int*   edge   = (const int*)d_in[1];
  const int*   src    = edge;
  const int*   dst    = edge + NEDGES;
  const int*   batch  = (const int*)d_in[2];
  const float* w1l = (const float*)d_in[3];  const float* b1 = (const float*)d_in[4];
  const float* w1r = (const float*)d_in[5];
  const float* w2l = (const float*)d_in[6];  const float* b2 = (const float*)d_in[7];
  const float* w2r = (const float*)d_in[8];
  const float* w3l = (const float*)d_in[9];  const float* b3 = (const float*)d_in[10];
  const float* w3r = (const float*)d_in[11];
  const float* fc1w = (const float*)d_in[12]; const float* fc1b_ = (const float*)d_in[13];
  const float* fc2w = (const float*)d_in[14]; const float* fc2b_ = (const float*)d_in[15];
  const float* clsw = (const float*)d_in[16]; const float* clsb_ = (const float*)d_in[17];

  // workspace carve-out (~246 MB; AGG fp32 aliases H3 region)
  char* ws = (char*)d_ws;
  size_t off = 0;
  auto alloc = [&](size_t bytes) -> void* {
    off = (off + 255) & ~(size_t)255;
    void* p = ws + off;
    off += bytes;
    return p;
  };
  __bf16* X0b  = (__bf16*)alloc((size_t)NNODES * 64 * 2);
  __bf16* H1b  = (__bf16*)alloc((size_t)NNODES * 128 * 2);
  __bf16* H2b  = (__bf16*)alloc((size_t)NNODES * 256 * 2);
  float*  AGG  = (float*)alloc((size_t)NNODES * 256 * 4);   // aliased with H3b
  __bf16* H3b  = (__bf16*)AGG;                              // bf16 NNODES*512 (same bytes)
  __bf16* AGGb = (__bf16*)alloc((size_t)NNODES * 256 * 2);
  float*  deg    = (float*)alloc((size_t)NNODES * 4);
  float*  invdeg = (float*)alloc((size_t)NNODES * 4);
  float*  pooled = (float*)alloc((size_t)NGRAPH * 512 * 4);
  float*  cnt    = (float*)alloc((size_t)NGRAPH * 4);
  __bf16* pooledb = (__bf16*)alloc((size_t)NGRAPH * 512 * 2);
  __bf16* z1b = (__bf16*)alloc((size_t)NGRAPH * 256 * 2);
  __bf16* z2b = (__bf16*)alloc((size_t)NGRAPH * 128 * 2);
  __bf16* w1lb = (__bf16*)alloc((size_t)128 * 64 * 2);
  __bf16* w1rb = (__bf16*)alloc((size_t)128 * 64 * 2);
  __bf16* w2lb = (__bf16*)alloc((size_t)256 * 128 * 2);
  __bf16* w2rb = (__bf16*)alloc((size_t)256 * 128 * 2);
  __bf16* w3lb = (__bf16*)alloc((size_t)512 * 256 * 2);
  __bf16* w3rb = (__bf16*)alloc((size_t)512 * 256 * 2);
  __bf16* fc1wb = (__bf16*)alloc((size_t)256 * 512 * 2);
  __bf16* fc2wb = (__bf16*)alloc((size_t)128 * 256 * 2);
  __bf16* clswb = (__bf16*)alloc((size_t)16 * 128 * 2);
  (void)ws_size; (void)n_in; (void)in_sizes; (void)out_size;

  const int T = 256;

  // degrees
  hipMemsetAsync(deg, 0, (size_t)NNODES * 4, stream);
  count_kernel<<<blks(NEDGES, T), T, 0, stream>>>(dst, deg, NEDGES);
  recip_clamp_kernel<<<blks(NNODES, T), T, 0, stream>>>(deg, invdeg, NNODES);

  // bf16 conversions (x padded 50->64, weights; cls padded 15->16 rows)
  cvt_pad_kernel<<<blks((long long)NNODES * 64, T), T, 0, stream>>>(x, X0b, NNODES, 64, NNODES, 50);
  cvt_pad_kernel<<<blks(128 * 64, T), T, 0, stream>>>(w1l, w1lb, 128, 64, 128, 50);
  cvt_pad_kernel<<<blks(128 * 64, T), T, 0, stream>>>(w1r, w1rb, 128, 64, 128, 50);
  cvt_pad_kernel<<<blks(256 * 128, T), T, 0, stream>>>(w2l, w2lb, 256, 128, 256, 128);
  cvt_pad_kernel<<<blks(256 * 128, T), T, 0, stream>>>(w2r, w2rb, 256, 128, 256, 128);
  cvt_pad_kernel<<<blks(512 * 256, T), T, 0, stream>>>(w3l, w3lb, 512, 256, 512, 256);
  cvt_pad_kernel<<<blks(512 * 256, T), T, 0, stream>>>(w3r, w3rb, 512, 256, 512, 256);
  cvt_pad_kernel<<<blks(256 * 512, T), T, 0, stream>>>(fc1w, fc1wb, 256, 512, 256, 512);
  cvt_pad_kernel<<<blks(128 * 256, T), T, 0, stream>>>(fc2w, fc2wb, 128, 256, 128, 256);
  cvt_pad_kernel<<<blks(16 * 128, T), T, 0, stream>>>(clsw, clswb, 16, 128, 15, 128);

  // NT=4: 16x64 strip per wave (N must be divisible by 64)
  auto gemm4 = [&](const __bf16* A1, const __bf16* B1, int K1,
                   const __bf16* A2, const __bf16* B2, int K2,
                   const float* bias, __bf16* Cb, float* Cf,
                   int M, int N, int ldc, int nvalid, int act) {
    long long waves = (long long)(M / 16) * (N / 64);
    wmma_gemm_kernel<4><<<blks(waves * 32, T), T, 0, stream>>>(
        A1, B1, K1, A2, B2, K2, bias, Cb, Cf, M, N, ldc, nvalid, act);
  };
  auto gemm1 = [&](const __bf16* A1, const __bf16* B1, int K1,
                   const float* bias, __bf16* Cb, float* Cf,
                   int M, int N, int ldc, int nvalid, int act) {
    long long waves = (long long)(M / 16) * (N / 16);
    wmma_gemm_kernel<1><<<blks(waves * 32, T), T, 0, stream>>>(
        A1, B1, K1, nullptr, nullptr, 0, bias, Cb, Cf, M, N, ldc, nvalid, act);
  };

  // ---- layer 1: K=64 (padded), N=128, leaky_relu(0.01) ----
  hipMemsetAsync(AGG, 0, (size_t)NNODES * 64 * 4, stream);
  scatter_kernel<<<blks((long long)NEDGES * 8, T), T, 0, stream>>>(X0b, src, dst, AGG, NEDGES, 64);
  scale_cvt_kernel<<<blks((long long)NNODES * 64, T), T, 0, stream>>>(AGG, invdeg, AGGb, (long long)NNODES * 64, 64);
  gemm4(AGGb, w1lb, 64, X0b, w1rb, 64, b1, H1b, nullptr, NNODES, 128, 128, 128, /*lrelu*/2);

  // ---- layer 2: K=128, N=256, relu ----
  hipMemsetAsync(AGG, 0, (size_t)NNODES * 128 * 4, stream);
  scatter_kernel<<<blks((long long)NEDGES * 16, T), T, 0, stream>>>(H1b, src, dst, AGG, NEDGES, 128);
  scale_cvt_kernel<<<blks((long long)NNODES * 128, T), T, 0, stream>>>(AGG, invdeg, AGGb, (long long)NNODES * 128, 128);
  gemm4(AGGb, w2lb, 128, H1b, w2rb, 128, b2, H2b, nullptr, NNODES, 256, 256, 256, 1);

  // ---- layer 3: K=256, N=512, relu (output aliases AGG region) ----
  hipMemsetAsync(AGG, 0, (size_t)NNODES * 256 * 4, stream);
  scatter_kernel<<<blks((long long)NEDGES * 32, T), T, 0, stream>>>(H2b, src, dst, AGG, NEDGES, 256);
  scale_cvt_kernel<<<blks((long long)NNODES * 256, T), T, 0, stream>>>(AGG, invdeg, AGGb, (long long)NNODES * 256, 256);
  gemm4(AGGb, w3lb, 256, H2b, w3rb, 256, b3, H3b, nullptr, NNODES, 512, 512, 512, 1);

  // ---- global mean pool ----
  hipMemsetAsync(pooled, 0, (size_t)NGRAPH * 512 * 4, stream);
  hipMemsetAsync(cnt, 0, (size_t)NGRAPH * 4, stream);
  count_kernel<<<blks(NNODES, T), T, 0, stream>>>(batch, cnt, NNODES);
  pool_scatter_kernel<<<blks((long long)NNODES * 64, T), T, 0, stream>>>(H3b, batch, pooled, NNODES, 512);
  pool_fin_kernel<<<blks((long long)NGRAPH * 512, T), T, 0, stream>>>(pooled, cnt, pooledb, NGRAPH, 512);

  // ---- FC head ----
  gemm4(pooledb, fc1wb, 512, nullptr, nullptr, 0, fc1b_, z1b, nullptr, NGRAPH, 256, 256, 256, 1);
  gemm4(z1b, fc2wb, 256, nullptr, nullptr, 0, fc2b_, z2b, nullptr, NGRAPH, 128, 128, 128, 1);
  // final: N padded to 16, valid 15, fp32 straight to d_out (ldc = 15)
  gemm1(z2b, clswb, 128, clsb_, nullptr, (float*)d_out, NGRAPH, 16, 15, 15, 0);
}
